// Attention_85796266705382
// MI455X (gfx1250) — matrix operand order repaired and checked by reading, not compiled, not run on hardware
//
#include <hip/hip_runtime.h>
#include <hip/hip_bf16.h>

typedef __attribute__((ext_vector_type(16))) __bf16        v16bf;
typedef __attribute__((ext_vector_type(8)))  float         v8f;
typedef __attribute__((ext_vector_type(4)))  unsigned int  u32x4;
typedef __attribute__((ext_vector_type(2)))  unsigned int  u32x2;
typedef __attribute__((ext_vector_type(4)))  float         f32x4;

union Frag {
    u32x4 q[2];
    v16bf bf;
};

#define ATTN_SCALE 0.35355339059327373f  /* b**-0.5 with b=8, per reference */

__device__ __forceinline__ unsigned short f2bf(float f) {
    unsigned int u = __builtin_bit_cast(unsigned int, f);
    u += 0x7FFFu + ((u >> 16) & 1u);   // round-to-nearest-even
    return (unsigned short)(u >> 16);
}

__device__ __forceinline__ v8f wmma_bf16(const Frag& a, const Frag& b, v8f c) {
    return __builtin_amdgcn_wmma_f32_16x16x32_bf16(false, a.bf, false, b.bf,
                                                   (short)0, c, false, false);
}

// gfx1250 async DMA: global memory -> LDS, tracked by ASYNCcnt.
__device__ __forceinline__ void async_load_b128(void* lds, const void* g) {
    unsigned l = (unsigned)(size_t)lds;                 // low 32 bits = LDS addr
    unsigned long long a = (unsigned long long)(size_t)g;
    asm volatile("global_load_async_to_lds_b128 %0, %1, off"
                 :: "v"(l), "v"(a) : "memory");
}
__device__ __forceinline__ void wait_async() {
    asm volatile("s_wait_asynccnt 0" ::: "memory");
}

// ---------------------------------------------------------------------------
// Kernel 1: QKV projection.  X (8192x512 f32) * Wqkv (512x1536 f32).
// Writes Q (pre-scaled) and K as (b,h,n,dh) bf16; V transposed as (b,h,dh,n).
// Block: 256 thr (8 waves), tile 128(M) x 64(N), K chunks of 32 staged in LDS.
// ---------------------------------------------------------------------------
__global__ __launch_bounds__(256) void qkv_gemm(
        const float* __restrict__ X, const float* __restrict__ W,
        unsigned short* __restrict__ qb, unsigned short* __restrict__ kb,
        unsigned short* __restrict__ vtb) {
    __shared__ unsigned short As[128 * 32];
    __shared__ unsigned short Bs[64 * 32];

    const int tid   = threadIdx.x;
    const int lane  = tid & 31;
    const int wid   = tid >> 5;
    const int waveM = wid & 3;
    const int waveN = wid >> 2;
    const int n0    = blockIdx.x * 64;    // N in [0,1536)
    const int m0    = blockIdx.y * 128;   // M in [0,8192)
    const int l15   = lane & 15;
    const int lh    = lane >> 4;
    const int kOffA = lh * 8;

    v8f c[2][2] = {};

    for (int k0 = 0; k0 < 512; k0 += 32) {
        // stage A tile: 128x32 f32 -> bf16
        #pragma unroll
        for (int it = 0; it < 4; ++it) {
            int i  = tid + it * 256;
            int r  = i >> 3;
            int kq = (i & 7) * 4;
            f32x4 f = *(const f32x4*)(X + (size_t)(m0 + r) * 512 + k0 + kq);
            unsigned int lo = (unsigned int)f2bf(f.x) | ((unsigned int)f2bf(f.y) << 16);
            unsigned int hi = (unsigned int)f2bf(f.z) | ((unsigned int)f2bf(f.w) << 16);
            u32x2 p; p.x = lo; p.y = hi;
            *(u32x2*)(&As[r * 32 + kq]) = p;
        }
        // stage B tile transposed: W[k0..k0+32][n0..n0+64] -> Bs[n][k]
        #pragma unroll
        for (int it = 0; it < 2; ++it) {
            int i  = tid + it * 256;
            int k  = i >> 4;
            int nq = (i & 15) * 4;
            f32x4 f = *(const f32x4*)(W + (size_t)(k0 + k) * 1536 + n0 + nq);
            Bs[(nq + 0) * 32 + k] = f2bf(f.x);
            Bs[(nq + 1) * 32 + k] = f2bf(f.y);
            Bs[(nq + 2) * 32 + k] = f2bf(f.z);
            Bs[(nq + 3) * 32 + k] = f2bf(f.w);
        }
        __syncthreads();

        Frag a[2], b[2];
        #pragma unroll
        for (int ti = 0; ti < 2; ++ti) {
            const unsigned short* ap = &As[(waveM * 32 + ti * 16 + l15) * 32];
            a[ti].q[0] = *(const u32x4*)(ap + kOffA);
            a[ti].q[1] = *(const u32x4*)(ap + 16 + kOffA);
        }
        #pragma unroll
        for (int tj = 0; tj < 2; ++tj) {
            const unsigned short* bp = &Bs[(waveN * 32 + tj * 16 + l15) * 32 + lh * 8];
            b[tj].q[0] = *(const u32x4*)(bp);
            b[tj].q[1] = *(const u32x4*)(bp + 16);
        }
        #pragma unroll
        for (int ti = 0; ti < 2; ++ti)
            #pragma unroll
            for (int tj = 0; tj < 2; ++tj)
                c[ti][tj] = wmma_bf16(a[ti], b[tj], c[ti][tj]);
        __syncthreads();
    }

    // scatter into q / k / v^T bf16 buffers.
    // part / head / d-base / batch are wave-uniform: force them scalar so the
    // q-k-v dispatch compiles to scalar branches instead of exec-mask juggling.
    const int bidx = __builtin_amdgcn_readfirstlane(m0 >> 10);
    #pragma unroll
    for (int tj = 0; tj < 2; ++tj) {
        const int nbase = __builtin_amdgcn_readfirstlane(n0 + waveN * 32 + tj * 16);
        const int part  = nbase >> 9;        // 0=q 1=k 2=v
        const int dbase = nbase & 511;
        const int hh    = dbase >> 6;
        const int d0    = dbase & 63;        // multiple of 16
        const size_t bhq = (size_t)((bidx << 3) + hh);
        #pragma unroll
        for (int ti = 0; ti < 2; ++ti) {
            const int tokb = (m0 & 1023) + waveM * 32 + ti * 16;
            #pragma unroll
            for (int v = 0; v < 8; ++v) {
                const int tok = tokb + v + lh * 8;
                const float val = c[ti][tj][v];
                if (part == 0)
                    qb[(bhq * 1024 + tok) * 64 + d0 + l15] = f2bf(val * ATTN_SCALE);
                else if (part == 1)
                    kb[(bhq * 1024 + tok) * 64 + d0 + l15] = f2bf(val);
                else
                    vtb[(bhq * 64 + d0 + l15) * 1024 + tok] = f2bf(val);
            }
        }
    }
}

// ---------------------------------------------------------------------------
// Kernel 2: flash attention, one (b,h,query-block-of-64) per workgroup.
// 4 waves x 16 query rows; keys streamed in chunks of 32 with K/V tiles
// cooperatively DMA'd into LDS via global_load_async_to_lds_b128.
// ---------------------------------------------------------------------------
__global__ __launch_bounds__(128) void attn_kernel(
        const unsigned short* __restrict__ qb, const unsigned short* __restrict__ kb,
        const unsigned short* __restrict__ vtb, unsigned short* __restrict__ ob) {
    __shared__ unsigned short Ks[32 * 64];     // (key, dh) tile
    __shared__ unsigned short Vs[64 * 32];     // (dh, key) tile
    __shared__ unsigned short Ps[4 * 16 * 32]; // per-wave P staging

    const int tid  = threadIdx.x;
    const int lane = tid & 31;
    const int w    = tid >> 5;
    const int bh   = blockIdx.x >> 4;
    const int qblk = blockIdx.x & 15;
    const int bidx = bh >> 3;
    const int h    = bh & 7;
    const int q0   = qblk * 64 + w * 16;
    const int l15  = lane & 15;
    const int lh   = lane >> 4;
    const int kOffA = lh * 8;

    // Q A-fragments for dh chunks [0,32) and [32,64)  (kept in registers)
    Frag aq[2];
    {
        const unsigned short* qp = qb + ((size_t)bh * 1024 + q0 + l15) * 64;
        aq[0].q[0] = *(const u32x4*)(qp + kOffA);
        aq[0].q[1] = *(const u32x4*)(qp + 16 + kOffA);
        aq[1].q[0] = *(const u32x4*)(qp + 32 + kOffA);
        aq[1].q[1] = *(const u32x4*)(qp + 48 + kOffA);
    }

    v8f o[4] = {};
    float mrow[8], lrow[8];
    #pragma unroll
    for (int v = 0; v < 8; ++v) { mrow[v] = -1e30f; lrow[v] = 0.0f; }

    const unsigned short* kbase = kb  + (size_t)bh * 1024 * 64;
    const unsigned short* vbase = vtb + (size_t)bh * 64 * 1024;
    unsigned short* myP = &Ps[w * 512];

    for (int j0 = 0; j0 < 1024; j0 += 32) {
        // ---- async-DMA K tile (contiguous 4KB) and V tile (64 rows x 64B) ----
        #pragma unroll
        for (int it = 0; it < 2; ++it) {
            int cidx = tid + it * 128;                 // 256 chunks of 16B
            async_load_b128(&Ks[cidx * 8], kbase + (size_t)j0 * 64 + cidx * 8);
        }
        #pragma unroll
        for (int it = 0; it < 2; ++it) {
            int cidx = tid + it * 128;
            int d  = cidx >> 2;
            int pt = (cidx & 3) * 8;
            async_load_b128(&Vs[d * 32 + pt], vbase + (size_t)d * 1024 + j0 + pt);
        }
        if (j0 + 32 < 1024)
            __builtin_prefetch(kbase + (size_t)(j0 + 32) * 64, 0, 1);
        wait_async();
        __syncthreads();

        // ---- S = Q K^T : two 16-key N-tiles, K=dh in 2 chunks of 32 ----
        v8f s[2] = {};
        #pragma unroll
        for (int nt = 0; nt < 2; ++nt) {
            const unsigned short* kp = &Ks[(nt * 16 + l15) * 64];
            Frag b0, b1;
            b0.q[0] = *(const u32x4*)(kp + lh * 8);
            b0.q[1] = *(const u32x4*)(kp + 16 + lh * 8);
            b1.q[0] = *(const u32x4*)(kp + 32 + lh * 8);
            b1.q[1] = *(const u32x4*)(kp + 48 + lh * 8);
            s[nt] = wmma_bf16(aq[0], b0, s[nt]);
            s[nt] = wmma_bf16(aq[1], b1, s[nt]);
        }
        // ---- online softmax over the 32 new columns ----
        float p0[8], p1[8];
        #pragma unroll
        for (int v = 0; v < 8; ++v) {
            float t = fmaxf(s[0][v], s[1][v]);
            #pragma unroll
            for (int off = 8; off; off >>= 1)
                t = fmaxf(t, __shfl_xor(t, off, 16));
            const float mn   = fmaxf(mrow[v], t);
            const float corr = __expf(mrow[v] - mn);
            mrow[v] = mn;
            const float e0 = __expf(s[0][v] - mn);
            const float e1 = __expf(s[1][v] - mn);
            p0[v] = e0; p1[v] = e1;
            float rs = e0 + e1;
            #pragma unroll
            for (int off = 8; off; off >>= 1)
                rs += __shfl_xor(rs, off, 16);
            lrow[v] = lrow[v] * corr + rs;
            o[0][v] *= corr; o[1][v] *= corr; o[2][v] *= corr; o[3][v] *= corr;
        }
        // ---- transpose P (C layout -> A layout) through per-wave LDS ----
        #pragma unroll
        for (int v = 0; v < 8; ++v) {
            const int r = v + lh * 8;
            myP[r * 32 + l15]      = f2bf(p0[v]);
            myP[r * 32 + 16 + l15] = f2bf(p1[v]);
        }
        Frag ap;
        {
            const unsigned short* pp = &myP[l15 * 32];
            ap.q[0] = *(const u32x4*)(pp + kOffA);
            ap.q[1] = *(const u32x4*)(pp + 16 + kOffA);
        }
        // ---- O += P V : 4 dh tiles of 16 ----
        #pragma unroll
        for (int dt = 0; dt < 4; ++dt) {
            const unsigned short* vp = &Vs[(dt * 16 + l15) * 32 + lh * 8];
            Frag bv;
            bv.q[0] = *(const u32x4*)(vp);
            bv.q[1] = *(const u32x4*)(vp + 16);
            o[dt] = wmma_bf16(ap, bv, o[dt]);
        }
        __syncthreads();   // protect Ks/Vs against next iteration's DMA
    }

    // normalize + store to attention-out buffer (8192 x 512) bf16
    #pragma unroll
    for (int v = 0; v < 8; ++v) {
        const float inv = 1.0f / lrow[v];
        const int tok   = q0 + v + lh * 8;
        unsigned short* orow = ob + ((size_t)bidx * 1024 + tok) * 512 + h * 64;
        #pragma unroll
        for (int dt = 0; dt < 4; ++dt)
            orow[dt * 16 + l15] = f2bf(o[dt][v] * inv);
    }
}

// ---------------------------------------------------------------------------
// Kernel 3: output projection.  O (8192x512 bf16) * Wout (512x512 f32) + bias.
// A tile staged via async DMA (already bf16); B converted f32->bf16 in VGPRs.
// ---------------------------------------------------------------------------
__global__ __launch_bounds__(256) void out_gemm(
        const unsigned short* __restrict__ A, const float* __restrict__ W,
        const float* __restrict__ bias, float* __restrict__ out) {
    __shared__ unsigned short As[128 * 32];
    __shared__ unsigned short Bs[64 * 32];

    const int tid   = threadIdx.x;
    const int lane  = tid & 31;
    const int wid   = tid >> 5;
    const int waveM = wid & 3;
    const int waveN = wid >> 2;
    const int n0    = blockIdx.x * 64;
    const int m0    = blockIdx.y * 128;
    const int l15   = lane & 15;
    const int lh    = lane >> 4;
    const int kOffA = lh * 8;

    v8f c[2][2] = {};

    for (int k0 = 0; k0 < 512; k0 += 32) {
        // stage A tile via async DMA: 128 rows x 64B
        #pragma unroll
        for (int it = 0; it < 2; ++it) {
            int cidx = tid + it * 256;               // 512 chunks of 16B
            int r  = cidx >> 2;
            int kq = (cidx & 3) * 8;
            async_load_b128(&As[r * 32 + kq],
                            A + (size_t)(m0 + r) * 512 + k0 + kq);
        }
        // stage B transposed with f32->bf16 conversion
        #pragma unroll
        for (int it = 0; it < 2; ++it) {
            int i  = tid + it * 256;
            int k  = i >> 4;
            int nq = (i & 15) * 4;
            f32x4 f = *(const f32x4*)(W + (size_t)(k0 + k) * 512 + n0 + nq);
            Bs[(nq + 0) * 32 + k] = f2bf(f.x);
            Bs[(nq + 1) * 32 + k] = f2bf(f.y);
            Bs[(nq + 2) * 32 + k] = f2bf(f.z);
            Bs[(nq + 3) * 32 + k] = f2bf(f.w);
        }
        wait_async();
        __syncthreads();

        Frag a[2], b[2];
        #pragma unroll
        for (int ti = 0; ti < 2; ++ti) {
            const unsigned short* ap = &As[(waveM * 32 + ti * 16 + l15) * 32];
            a[ti].q[0] = *(const u32x4*)(ap + kOffA);
            a[ti].q[1] = *(const u32x4*)(ap + 16 + kOffA);
        }
        #pragma unroll
        for (int tj = 0; tj < 2; ++tj) {
            const unsigned short* bp = &Bs[(waveN * 32 + tj * 16 + l15) * 32 + lh * 8];
            b[tj].q[0] = *(const u32x4*)(bp);
            b[tj].q[1] = *(const u32x4*)(bp + 16);
        }
        #pragma unroll
        for (int ti = 0; ti < 2; ++ti)
            #pragma unroll
            for (int tj = 0; tj < 2; ++tj)
                c[ti][tj] = wmma_bf16(a[ti], b[tj], c[ti][tj]);
        __syncthreads();
    }

    #pragma unroll
    for (int tj = 0; tj < 2; ++tj) {
        const int n = n0 + waveN * 32 + tj * 16 + l15;
        const float bv = bias[n];
        #pragma unroll
        for (int ti = 0; ti < 2; ++ti) {
            #pragma unroll
            for (int v = 0; v < 8; ++v) {
                const int m = m0 + waveM * 32 + ti * 16 + v + lh * 8;
                out[(size_t)m * 512 + n] = c[ti][tj][v] + bv;
            }
        }
    }
}

// ---------------------------------------------------------------------------
extern "C" void kernel_launch(void* const* d_in, const int* in_sizes, int n_in,
                              void* d_out, int out_size, void* d_ws, size_t ws_size,
                              hipStream_t stream) {
    (void)in_sizes; (void)n_in; (void)out_size; (void)ws_size;
    const float* x     = (const float*)d_in[0];
    const float* w_qkv = (const float*)d_in[1];
    const float* w_out = (const float*)d_in[2];
    const float* b_out = (const float*)d_in[3];
    float* out = (float*)d_out;

    // workspace layout (bf16 = unsigned short): q, k, v^T, attn-out, each 4M elems
    unsigned short* qb  = (unsigned short*)d_ws;
    unsigned short* kb  = qb  + (size_t)4194304;
    unsigned short* vtb = kb  + (size_t)4194304;
    unsigned short* ob  = vtb + (size_t)4194304;

    qkv_gemm<<<dim3(24, 64), 256, 0, stream>>>(x, w_qkv, qb, kb, vtb);
    attn_kernel<<<1024, 128, 0, stream>>>(qb, kb, vtb, ob);
    out_gemm<<<dim3(8, 64), 256, 0, stream>>>(ob, w_out, b_out, out);
}
